// GNNModel_68427418959951
// MI455X (gfx1250) — matrix operand order repaired and checked by reading, not compiled, not run on hardware
//
#include <hip/hip_runtime.h>
#include <hip/hip_bf16.h>

typedef _Float16 v16h __attribute__((ext_vector_type(16)));
typedef _Float16 v8h  __attribute__((ext_vector_type(8)));
typedef float    v8f  __attribute__((ext_vector_type(8)));

#define D_FEAT 128

// ---------------- elementwise helpers ----------------

__global__ void k_deg_init(float* __restrict__ deg, int N) {
    int i = blockIdx.x * blockDim.x + threadIdx.x;
    if (i < N) deg[i] = 1.0f;   // self-loop contribution
}

__global__ void k_deg_edges(const int* __restrict__ dst, float* __restrict__ deg, int E) {
    int e = blockIdx.x * blockDim.x + threadIdx.x;
    if (e < E)
        __hip_atomic_fetch_add(&deg[dst[e]], 1.0f,
                               __ATOMIC_RELAXED, __HIP_MEMORY_SCOPE_AGENT);
}

__global__ void k_dinv(const float* __restrict__ deg, float* __restrict__ dinv, int N) {
    int i = blockIdx.x * blockDim.x + threadIdx.x;
    if (i < N) {
        float d = deg[i];
        dinv[i] = (d > 0.0f) ? rsqrtf(d) : 0.0f;
    }
}

__global__ void k_f32_to_f16(const float* __restrict__ in, _Float16* __restrict__ out, int n) {
    int i = blockIdx.x * blockDim.x + threadIdx.x;
    if (i < n) out[i] = (_Float16)in[i];
}

// agg[i,f] = bias[f] + h[i,f] * dinv[i]^2   (self-loop folded in)
__global__ void k_init_agg(const float* __restrict__ h, const float* __restrict__ dinv,
                           const float* __restrict__ bias, float* __restrict__ agg, int total) {
    int i = blockIdx.x * blockDim.x + threadIdx.x;
    if (i >= total) return;
    int node = i >> 7;
    int f    = i & (D_FEAT - 1);
    float d = dinv[node];
    agg[i] = bias[f] + h[i] * d * d;
}

// one lane per (edge, feature): coalesced gather + hw fp32 atomic scatter
__global__ __launch_bounds__(256)
void k_scatter(const int* __restrict__ src, const int* __restrict__ dst,
               const float* __restrict__ dinv, const float* __restrict__ h,
               float* __restrict__ agg, int E) {
    int idx = blockIdx.x * blockDim.x + threadIdx.x;
    int e = idx >> 7;
    int f = idx & (D_FEAT - 1);
    if (e >= E) return;
    int s = src[e];
    int d = dst[e];
    float w = dinv[s] * dinv[d];
    float v = h[(size_t)s * D_FEAT + f] * w;
    __hip_atomic_fetch_add(&agg[(size_t)d * D_FEAT + f], v,
                           __ATOMIC_RELAXED, __HIP_MEMORY_SCOPE_AGENT);
}

__global__ void k_relu_f16(const float* __restrict__ in, _Float16* __restrict__ out, int n) {
    int i = blockIdx.x * blockDim.x + threadIdx.x;
    if (i < n) out[i] = (_Float16)fmaxf(in[i], 0.0f);
}

__global__ void k_relu_f32(const float* __restrict__ in, float* __restrict__ out, int n) {
    int i = blockIdx.x * blockDim.x + threadIdx.x;
    if (i < n) out[i] = fmaxf(in[i], 0.0f);
}

// ---------------- weight pre-pack for WMMA B fragments ----------------
// B fragment k order used here (contiguous, same as the A loads): lane = (col&15) + 16*kh,
// element e (0..15) holds W[t*32 + kh*16 + e][col]. Packed so each lane's
// fragment is 16 contiguous halves (32B).
__global__ void k_pack_w(const float* __restrict__ W, _Float16* __restrict__ Wp) {
    int id = blockIdx.x * blockDim.x + threadIdx.x;   // 0 .. 16383
    if (id >= 16384) return;
    int e    = id & 15;
    int lane = (id >> 4) & 31;
    int n    = (id >> 9) & 7;
    int t    = id >> 12;
    int k    = t * 32 + (lane >> 4) * 16 + e;
    int col  = n * 16 + (lane & 15);
    Wp[id] = (_Float16)W[k * D_FEAT + col];
}

// ---------------- WMMA GEMM: C[N,128] = A[N,128] x B[128,128] ----------------
// One wave computes a 16x128 row block: 8 f32 accumulators, 4 K-steps,
// 32 x v_wmma_f32_16x16x32_f16 per wave.
__global__ __launch_bounds__(256)
void k_gemm_wmma(const _Float16* __restrict__ A, const _Float16* __restrict__ Bp,
                 float* __restrict__ C, int mtiles) {
    int wave = (blockIdx.x * blockDim.x + threadIdx.x) >> 5;
    int lane = threadIdx.x & 31;
    if (wave >= mtiles) return;                // wave-uniform: EXEC stays all-1s for WMMA

    const int m  = lane & 15;
    const int kh = lane >> 4;
    const _Float16* arow = A + (size_t)(wave * 16 + m) * D_FEAT;

    v8f acc[8] = {};

#pragma unroll
    for (int t = 0; t < 4; ++t) {
        // A fragment: 16x32 f16, two contiguous 16B loads per lane
        v8h alo = *(const v8h*)(arow + t * 32 + kh * 16);
        v8h ahi = *(const v8h*)(arow + t * 32 + kh * 16 + 8);
        v16h a;
#pragma unroll
        for (int i = 0; i < 8; ++i) { a[i] = alo[i]; a[8 + i] = ahi[i]; }

#pragma unroll
        for (int n = 0; n < 8; ++n) {
            v16h b = *(const v16h*)(Bp + ((size_t)(t * 8 + n) * 32 + lane) * 16);
            acc[n] = __builtin_amdgcn_wmma_f32_16x16x32_f16(
                false, a, false, b, (short)0, acc[n], false, false);
        }
    }

    // C/D layout: lanes 0-15 -> rows M=0..7 (vgpr r), lanes 16-31 -> rows M=8..15
    int rbase = wave * 16 + kh * 8;
    int col0  = lane & 15;
#pragma unroll
    for (int n = 0; n < 8; ++n)
#pragma unroll
        for (int r = 0; r < 8; ++r)
            C[(size_t)(rbase + r) * D_FEAT + n * 16 + col0] = acc[n][r];
}

// ---------------- FC head: out[n,0:2] = h[n,:] @ Wfc + bfc ----------------
__global__ void k_fc(const float* __restrict__ h, const float* __restrict__ Wfc,
                     const float* __restrict__ bfc, float* __restrict__ out, int N) {
    int n = blockIdx.x * blockDim.x + threadIdx.x;
    if (n >= N) return;
    const float* row = h + (size_t)n * D_FEAT;
    float a0 = bfc[0], a1 = bfc[1];
#pragma unroll 8
    for (int k = 0; k < D_FEAT; ++k) {
        float v = row[k];
        a0 += v * Wfc[2 * k];
        a1 += v * Wfc[2 * k + 1];
    }
    out[2 * n]     = a0;
    out[2 * n + 1] = a1;
}

// ---------------- host orchestration ----------------

static inline size_t align256(size_t x) { return (x + 255) & ~(size_t)255; }

extern "C" void kernel_launch(void* const* d_in, const int* in_sizes, int n_in,
                              void* d_out, int out_size, void* d_ws, size_t ws_size,
                              hipStream_t stream) {
    const float* x   = (const float*)d_in[0];
    const int*   ei  = (const int*)d_in[1];
    const float* W1  = (const float*)d_in[2];
    const float* b1  = (const float*)d_in[3];
    const float* W2  = (const float*)d_in[4];
    const float* b2  = (const float*)d_in[5];
    const float* Wfc = (const float*)d_in[6];
    const float* bfc = (const float*)d_in[7];
    float* out = (float*)d_out;

    const int N = in_sizes[0] / D_FEAT;
    const int E = in_sizes[1] / 2;
    const int* src = ei;
    const int* dst = ei + E;

    const int NF = N * D_FEAT;

    // workspace carve-up
    char* ws = (char*)d_ws;
    size_t off = 0;
    float*    deg  = (float*)(ws + off);  off = align256(off + (size_t)N * 4);
    float*    dinv = (float*)(ws + off);  off = align256(off + (size_t)N * 4);
    float*    hbuf = (float*)(ws + off);  off = align256(off + (size_t)NF * 4);
    float*    agg  = (float*)(ws + off);  off = align256(off + (size_t)NF * 4);
    _Float16* xh   = (_Float16*)(ws + off); off = align256(off + (size_t)NF * 2);
    _Float16* W1p  = (_Float16*)(ws + off); off = align256(off + 16384 * 2);
    _Float16* W2p  = (_Float16*)(ws + off); off = align256(off + 16384 * 2);
    (void)ws_size; (void)n_in; (void)out_size;

    const int TB = 256;
    const int gN   = (N  + TB - 1) / TB;
    const int gNF  = (NF + TB - 1) / TB;
    const int gE   = (E  + TB - 1) / TB;
    const int gEF  = (int)(((size_t)E * D_FEAT + TB - 1) / TB);
    const int mtiles = N / 16;                       // N = 50000 -> 3125 exact
    const int gGemm  = (mtiles + (TB / 32) - 1) / (TB / 32);

    // normalization
    k_deg_init<<<gN, TB, 0, stream>>>(deg, N);
    k_deg_edges<<<gE, TB, 0, stream>>>(dst, deg, E);
    k_dinv<<<gN, TB, 0, stream>>>(deg, dinv, N);

    // weight pre-pack (once per launch; deterministic)
    k_pack_w<<<64, TB, 0, stream>>>(W1, W1p);
    k_pack_w<<<64, TB, 0, stream>>>(W2, W2p);

    // layer 1
    k_f32_to_f16<<<gNF, TB, 0, stream>>>(x, xh, NF);
    k_gemm_wmma<<<gGemm, TB, 0, stream>>>(xh, W1p, hbuf, mtiles);
    k_init_agg<<<gNF, TB, 0, stream>>>(hbuf, dinv, b1, agg, NF);
    k_scatter<<<gEF, TB, 0, stream>>>(src, dst, dinv, hbuf, agg, E);
    k_relu_f16<<<gNF, TB, 0, stream>>>(agg, xh, NF);

    // layer 2
    k_gemm_wmma<<<gGemm, TB, 0, stream>>>(xh, W2p, hbuf, mtiles);
    k_init_agg<<<gNF, TB, 0, stream>>>(hbuf, dinv, b2, agg, NF);
    k_scatter<<<gEF, TB, 0, stream>>>(src, dst, dinv, hbuf, agg, E);
    k_relu_f32<<<gNF, TB, 0, stream>>>(agg, hbuf, NF);

    // FC head
    k_fc<<<gN, TB, 0, stream>>>(hbuf, Wfc, bfc, out, N);
}
